// CARAFE_39994735460624
// MI455X (gfx1250) — matrix operand order repaired and checked by reading, not compiled, not run on hardware
//
#include <hip/hip_runtime.h>
#include <hip/hip_bf16.h>

typedef __attribute__((ext_vector_type(16))) __bf16 v16bf;
typedef __attribute__((ext_vector_type(8)))  float  v8f;

// Native f32 -> bf16 convert; clang packs pairs into v_cvt_pk_bf16_f32.
__device__ __forceinline__ __bf16 f2bf(float f) { return (__bf16)f; }

// A-fragment, contiguous k order (B strips match): lane holds row m = lane%16;
// element i holds k = (lane/16)*16 + i  ->  two contiguous float4x2 runs.
__device__ __forceinline__ v16bf load_afrag(const float* __restrict__ arow, int h) {
    const float4* q0 = (const float4*)(arow + h * 16);
    const float4* q1 = (const float4*)(arow + h * 16 + 8);
    float4 x0 = q0[0], x1 = q0[1], y0 = q1[0], y1 = q1[1];
    v16bf a;
    a[0]  = f2bf(x0.x); a[1]  = f2bf(x0.y); a[2]  = f2bf(x0.z); a[3]  = f2bf(x0.w);
    a[4]  = f2bf(x1.x); a[5]  = f2bf(x1.y); a[6]  = f2bf(x1.z); a[7]  = f2bf(x1.w);
    a[8]  = f2bf(y0.x); a[9]  = f2bf(y0.y); a[10] = f2bf(y0.z); a[11] = f2bf(y0.w);
    a[12] = f2bf(y1.x); a[13] = f2bf(y1.y); a[14] = f2bf(y1.z); a[15] = f2bf(y1.w);
    return a;
}

#define WMMA_BF16(A_, B_, C_) \
    __builtin_amdgcn_wmma_f32_16x16x32_bf16(false, (A_), false, (B_), (short)0, (C_), false, false)

// ---------------------------------------------------------------------------
// 1x1-conv GEMM:  C[n][m][p] = sum_k A[m][k] * B[n][k][p] + bias[m]
// One wave per 32x32 macro-tile: 2 A-frags x 2 B-frags -> 4 WMMAs per K-step.
// Compile-time M,K,P -> strided B loads are one base + immediate offsets.
// ---------------------------------------------------------------------------
template <int M, int K, int P>
__global__ void gemm1x1_wmma(const float* __restrict__ A,
                             const float* __restrict__ B,
                             const float* __restrict__ bias,
                             float* __restrict__ C,
                             int nb)
{
    constexpr int mt2 = M >> 5, pt2 = P >> 5;
    int wave = (int)((blockIdx.x * blockDim.x + threadIdx.x) >> 5);
    int lane = threadIdx.x & 31;
    int total = nb * mt2 * pt2;
    if (wave >= total) return;                 // wave-uniform: EXEC all-1s at WMMA

    int n  = wave / (mt2 * pt2);
    int rm = wave - n * (mt2 * pt2);
    int mi = rm / pt2;
    int pi = rm - mi * pt2;

    const float* Bn = B + (size_t)n * K * P;
    int l15 = lane & 15, h = lane >> 4;
    int m0 = mi * 32 + l15;
    int p0 = pi * 32 + l15;

    v8f acc00 = {}, acc01 = {}, acc10 = {}, acc11 = {};
    for (int kb = 0; kb < K; kb += 32) {
        v16bf a0 = load_afrag(A + (size_t)m0 * K + kb, h);
        v16bf a1 = load_afrag(A + (size_t)(m0 + 16) * K + kb, h);
        const float* bbase = Bn + (size_t)(kb + h * 16) * P + p0;
        v16bf b0, b1;
#pragma unroll
        for (int i = 0; i < 16; ++i) {         // immediate-offset strided loads
            b0[i] = f2bf(bbase[i * P]);
            b1[i] = f2bf(bbase[i * P + 16]);
        }
        if (kb + 32 < K)                       // next slab -> global_prefetch_b8
            __builtin_prefetch(bbase + (size_t)32 * P, 0, 3);
        acc00 = WMMA_BF16(a0, b0, acc00);
        acc01 = WMMA_BF16(a0, b1, acc01);
        acc10 = WMMA_BF16(a1, b0, acc10);
        acc11 = WMMA_BF16(a1, b1, acc11);
    }

    float* Cn = C + (size_t)n * M * P;
#pragma unroll
    for (int r = 0; r < 8; ++r) {
        int row0 = mi * 32 + h * 8 + r;        // C/D layout: m = r + (lane/16)*8
        int row1 = row0 + 16;
        float bv0 = bias[row0], bv1 = bias[row1];
        Cn[(size_t)row0 * P + p0]      = acc00[r] + bv0;
        Cn[(size_t)row0 * P + p0 + 16] = acc01[r] + bv0;
        Cn[(size_t)row1 * P + p0]      = acc10[r] + bv1;
        Cn[(size_t)row1 * P + p0 + 16] = acc11[r] + bv1;
    }
}

// ---------------------------------------------------------------------------
// One-time weight reorder for the encoder conv: tap-major reduction axis.
//   wt[m][tap*64 + cd] = w_enc[m][cd*9 + tap]
// ---------------------------------------------------------------------------
__global__ void reorder_wenc(const float* __restrict__ w, float* __restrict__ wt) {
    int idx = (int)blockIdx.x * 256 + (int)threadIdx.x;      // 100*576 = 57600
    if (idx >= 100 * 576) return;
    int m = idx / 576, k = idx - m * 576;
    int tap = k >> 6, cd = k & 63;
    wt[idx] = w[(size_t)m * 576 + cd * 9 + tap];
}

// ---------------------------------------------------------------------------
// 3x3 encoder conv as implicit GEMM, pad=1, tap-major K (k = tap*64 + cd).
// M = 100 (padded to 128), K = 576, P = 4096, nb = 4.
// Within a 16-element B strip: (kb + h*16) % 64 in {0,16,32,48} and i < 16,
// so tap / (dy,dx) / bounds are UNIFORM per strip and cd is contiguous ->
// one base + immediate offsets, one clamp-and-select per strip. No branches.
// ---------------------------------------------------------------------------
__global__ void enc3x3_wmma(const float* __restrict__ A,    // wt [100][576] tap-major
                            const float* __restrict__ B,    // t0 [4][64][4096]
                            const float* __restrict__ bias, // [100]
                            float* __restrict__ C)          // e  [4][100][4096]
{
    constexpr int M = 100, K = 576, P = 4096;
    constexpr int mt2 = 4, pt2 = 128;
    int wave = (int)((blockIdx.x * blockDim.x + threadIdx.x) >> 5);
    int lane = threadIdx.x & 31;
    int total = 4 * mt2 * pt2;
    if (wave >= total) return;

    int n  = wave / (mt2 * pt2);
    int rm = wave - n * (mt2 * pt2);
    int mi = rm / pt2;
    int pi = rm - mi * pt2;

    int l15 = lane & 15, h = lane >> 4;
    int m0 = mi * 32 + l15, m1 = m0 + 16;
    int mc0 = (m0 < M) ? m0 : (M - 1);         // clamp addr; padded rows discarded
    int mc1 = (m1 < M) ? m1 : (M - 1);
    int p0 = pi * 32 + l15, p1 = p0 + 16;
    int py0 = p0 >> 6, px0 = p0 & 63;
    int py1 = p1 >> 6, px1 = p1 & 63;

    const float* Bn = B + (size_t)n * 64 * P;

    v8f acc00 = {}, acc01 = {}, acc10 = {}, acc11 = {};
#pragma unroll 3
    for (int kb = 0; kb < K; kb += 32) {
        v16bf a0 = load_afrag(A + (size_t)mc0 * K + kb, h);
        v16bf a1 = load_afrag(A + (size_t)mc1 * K + kb, h);

        int kh  = kb + h * 16;                 // multiple of 16
        int tap = kh >> 6;                     // uniform over the strip
        int cd0 = kh & 63;                     // strip-contiguous channel base
        int dy  = tap / 3 - 1, dx = tap - (tap / 3) * 3 - 1;
        int yy0 = py0 + dy, xx0 = px0 + dx;
        int yy1 = py1 + dy, xx1 = px1 + dx;
        bool in0 = ((unsigned)yy0 < 64u) & ((unsigned)xx0 < 64u);
        bool in1 = ((unsigned)yy1 < 64u) & ((unsigned)xx1 < 64u);
        // wrapped address stays inside this batch's [64][4096] plane
        const float* s0 = Bn + (size_t)cd0 * P + ((yy0 & 63) * 64 + (xx0 & 63));
        const float* s1 = Bn + (size_t)cd0 * P + ((yy1 & 63) * 64 + (xx1 & 63));

        v16bf b0, b1;
#pragma unroll
        for (int i = 0; i < 16; ++i) {         // immediate-offset loads + cndmask
            b0[i] = f2bf(in0 ? s0[i * P] : 0.f);
            b1[i] = f2bf(in1 ? s1[i * P] : 0.f);
        }
        acc00 = WMMA_BF16(a0, b0, acc00);
        acc01 = WMMA_BF16(a0, b1, acc01);
        acc10 = WMMA_BF16(a1, b0, acc10);
        acc11 = WMMA_BF16(a1, b1, acc11);
    }

    float* Cn = C + (size_t)n * M * P;
#pragma unroll
    for (int r = 0; r < 8; ++r) {
        int row0 = mi * 32 + h * 8 + r;
        int row1 = row0 + 16;
        if (row0 < M) {
            float bv = bias[row0];
            Cn[(size_t)row0 * P + p0] = acc00[r] + bv;
            Cn[(size_t)row0 * P + p1] = acc01[r] + bv;
        }
        if (row1 < M) {
            float bv = bias[row1];
            Cn[(size_t)row1 * P + p0] = acc10[r] + bv;
            Cn[(size_t)row1 * P + p1] = acc11[r] + bv;
        }
    }
}

// ---------------------------------------------------------------------------
// Pixel-shuffle + softmax + block-unfold fused:
//   kt[n][h][kk][si][w] = softmax_kk( e[n][kk*4 + si][h][w] )
// ---------------------------------------------------------------------------
__global__ void softmax_kt(const float* __restrict__ e,   // [4][100][4096]
                           float* __restrict__ kt)        // [4][64][25][4][64]
{
    int n    = blockIdx.x >> 6;
    int hrow = blockIdx.x & 63;
    int si   = threadIdx.x >> 6;
    int w    = threadIdx.x & 63;

    const float* ep = e + ((size_t)n * 100 + si) * 4096 + hrow * 64 + w;
    float v[25];
    float mx = -3.4e38f;
#pragma unroll
    for (int kk = 0; kk < 25; ++kk) {
        v[kk] = ep[(size_t)kk * 4 * 4096];
        mx = fmaxf(mx, v[kk]);
    }
    float s = 0.f;
#pragma unroll
    for (int kk = 0; kk < 25; ++kk) { v[kk] = __expf(v[kk] - mx); s += v[kk]; }
    float inv = 1.f / s;

    float* kp = kt + ((((size_t)n * 64 + hrow) * 25) * 4 + si) * 64 + w;
#pragma unroll
    for (int kk = 0; kk < 25; ++kk)
        kp[(size_t)kk * 4 * 64] = v[kk] * inv;
}

// ---------------------------------------------------------------------------
// Content-aware reassembly + pixel shuffle (K=25 -> VALU FMA, coalesced in w).
// Clamp-and-select instead of branches for the halo.
// ---------------------------------------------------------------------------
__global__ void reassemble(const float* __restrict__ x,   // [4][256][64][64]
                           const float* __restrict__ kt,  // [4][64][25][4][64]
                           float* __restrict__ y)         // [4][256][128][128]
{
    int n    = (int)blockIdx.x >> 6;
    int hrow = (int)blockIdx.x & 63;
    int cq   = threadIdx.x >> 6;
    int w    = threadIdx.x & 63;
    int c    = (int)blockIdx.y * 4 + cq;

    const float* xp = x + ((size_t)n * 256 + c) * 4096;
    const float* kp = kt + ((size_t)n * 64 + hrow) * 25 * 4 * 64 + w;

    float a0 = 0.f, a1 = 0.f, a2 = 0.f, a3 = 0.f;
#pragma unroll
    for (int kk = 0; kk < 25; ++kk) {
        int dy = kk / 5 - 2, dx = kk % 5 - 2;
        int yy = hrow + dy, xx = w + dx;
        float xv = xp[(yy & 63) * 64 + (xx & 63)];          // wrapped, always valid
        xv = (((unsigned)yy < 64u) & ((unsigned)xx < 64u)) ? xv : 0.f;
        const float* kq = kp + (size_t)kk * 256;
        a0 = fmaf(xv, kq[0],   a0);
        a1 = fmaf(xv, kq[64],  a1);
        a2 = fmaf(xv, kq[128], a2);
        a3 = fmaf(xv, kq[192], a3);
    }
    float* yp = y + (((size_t)n * 256 + c) * 128 + 2 * hrow) * 128 + 2 * w;
    yp[0]   = a0;  yp[1]   = a1;
    yp[128] = a2;  yp[129] = a3;
}

// ---------------------------------------------------------------------------
extern "C" void kernel_launch(void* const* d_in, const int* in_sizes, int n_in,
                              void* d_out, int out_size, void* d_ws, size_t ws_size,
                              hipStream_t stream) {
    (void)in_sizes; (void)n_in; (void)out_size; (void)ws_size;

    const float* x      = (const float*)d_in[0];   // [4,256,64,64]
    const float* w_down = (const float*)d_in[1];   // [64,256,1,1]
    const float* b_down = (const float*)d_in[2];   // [64]
    const float* w_enc  = (const float*)d_in[3];   // [100,64,3,3]
    const float* b_enc  = (const float*)d_in[4];   // [100]
    const float* w_out  = (const float*)d_in[5];   // [256,256,1,1]
    const float* b_out  = (const float*)d_in[6];   // [256]
    float* out = (float*)d_out;                    // [4,256,128,128]

    float* t0 = (float*)d_ws;                                  // [4][64][4096]
    float* e  = t0 + (size_t)4 * 64  * 4096;                   // [4][100][4096]
    float* kt = e  + (size_t)4 * 100 * 4096;                   // [4][64][25][4][64]
    float* y  = kt + (size_t)4 * 64  * 25 * 4 * 64;            // [4][256][16384]
    float* wt = y  + (size_t)4 * 256 * 16384;                  // [100][576] tap-major

    // weight reorder for encoder (tiny, once per launch)
    reorder_wenc<<<225, 256, 0, stream>>>(w_enc, wt);

    // 1x1 down conv: waves = 4 * (64/32) * (4096/32) = 1024 -> 128 blocks of 8 waves
    gemm1x1_wmma<64, 256, 4096><<<128, 256, 0, stream>>>(w_down, x, b_down, t0, 4);

    // 3x3 encoder conv: waves = 4 * 4 * 128 = 2048 -> 256 blocks
    enc3x3_wmma<<<256, 256, 0, stream>>>(wt, t0, b_enc, e);

    // softmax over k^2 = 25
    softmax_kt<<<256, 256, 0, stream>>>(e, kt);

    // reassembly + pixel shuffle
    reassemble<<<dim3(256, 64), 256, 0, stream>>>(x, kt, y);

    // 1x1 out conv: waves = 4 * (256/32) * (16384/32) = 16384 -> 2048 blocks
    gemm1x1_wmma<256, 256, 16384><<<2048, 256, 0, stream>>>(w_out, y, b_out, out, 4);
}
